// RXN_Sequence_43198781063730
// MI455X (gfx1250) — compile-verified, matrix operand order checked by reading, not run on hardware
//
#include <hip/hip_runtime.h>
#include <hip/hip_bf16.h>
#include <math.h>

// ---------------------------------------------------------------------------
// Problem constants (from the reference)
// ---------------------------------------------------------------------------
#define ELn   8
#define Bg    32
#define Nn    64
#define An    (Bg * Nn)          // 2048 nodes
#define NAa   98
#define NAp   128                // padded K for node features
#define NBb   6
#define NBp   64                 // padded K for edge features
#define Dd    256
#define Hh    8
#define DH    32                 // Dd / Hh
#define En    (Bg * 256)         // 8192 edges
#define Pp    (Nn * (Nn - 1) / 2)// 2016 pairs per graph
#define BPn   (Bg * Pp)          // 64512
#define VOCn  5

#define LDSTRIDE 66              // padded LDS row stride (floats) - no bank conflicts

typedef __attribute__((ext_vector_type(16))) _Float16 v16h;
typedef __attribute__((ext_vector_type(8)))  float    v8f;
typedef __attribute__((address_space(3)))    float    lds_f32;

// ---------------------------------------------------------------------------
// Fragment loaders (CDNA5 ISA §7.12.2 layouts, wave32)
// A 16x32 f16: lane(l) -> row = l&15; hi = l>>4 selects K sub-blocks.
// Lane needs K = hi*8 + 0..7 and 16 + hi*8 + 0..7  ==> two contiguous
// 8-float runs of one row ==> 4x global_load_b128.
// ---------------------------------------------------------------------------
__device__ __forceinline__ v16h load_afrag(const float* __restrict__ p, int hi)
{
    const float4* q0 = (const float4*)(p + hi * 8);
    const float4* q1 = (const float4*)(p + 16 + hi * 8);
    float4 f0 = q0[0], f1 = q0[1], f2 = q1[0], f3 = q1[1];
    v16h a;
    a[0]  = (_Float16)f0.x; a[1]  = (_Float16)f0.y; a[2]  = (_Float16)f0.z; a[3]  = (_Float16)f0.w;
    a[4]  = (_Float16)f1.x; a[5]  = (_Float16)f1.y; a[6]  = (_Float16)f1.z; a[7]  = (_Float16)f1.w;
    a[8]  = (_Float16)f2.x; a[9]  = (_Float16)f2.y; a[10] = (_Float16)f2.z; a[11] = (_Float16)f2.w;
    a[12] = (_Float16)f3.x; a[13] = (_Float16)f3.y; a[14] = (_Float16)f3.z; a[15] = (_Float16)f3.w;
    return a;
}

// B fragment from LDS tile staged transposed as shB[col][k] (stride LDSTRIDE).
// p points at shB + col*LDSTRIDE; kk even -> 8-byte aligned ds_load_b64 pairs.
template <bool GUARD>
__device__ __forceinline__ v16h load_bfrag_lds(const float* __restrict__ p, int hi,
                                               int koff, float sc)
{
    v16h b;
#pragma unroll
    for (int v = 0; v < 8; ++v) {
        int kk = koff + ((v < 4) ? (hi * 8 + 2 * v) : (16 + hi * 8 + 2 * (v - 4)));
        float2 f = *(const float2*)(p + kk);
        float b0 = f.x, b1 = f.y;
        if (GUARD) { b0 *= sc; b1 *= sc; }
        b[2 * v]     = (_Float16)b0;
        b[2 * v + 1] = (_Float16)b1;
    }
    return b;
}

// ---------------------------------------------------------------------------
// WMMA GEMM:  C[M,N] = act( A[M,K] @ W[K,N] + bias + resid )
// Requirements: M % 16 == 0, K % 64 == 0 (inputs padded by host code).
// Block = 128 threads (4 waves) computing a 64x32 macro-tile; the shared
// 64x32 B tile is staged into LDS (transposed) per K-block via the CDNA5
// async global->LDS engine (ASYNCcnt), then each wave runs 4 WMMAs per
// K-block on fragments read with ds_load_b64.  NGUARD=false when N%32==0.
// ---------------------------------------------------------------------------
template <bool NGUARD>
__global__ __launch_bounds__(128)
void k_gemm_wmma(const float* __restrict__ A, const float* __restrict__ W,
                 const float* __restrict__ bias, const float* __restrict__ resid,
                 float* __restrict__ C, int M, int K, int N, int relu)
{
    __shared__ float shB[32 * LDSTRIDE];
    int tilesN = (N + 31) >> 5;
    int bm = (blockIdx.x / tilesN) << 6;     // 64-row macro tile
    int tn = (blockIdx.x % tilesN) << 5;     // 32-col tile
    int tid  = threadIdx.x;
    int wave = tid >> 5;
    int lane = tid & 31;
    int r  = lane & 15;
    int hi = lane >> 4;
    int tm = bm + (wave << 4);
    bool rowsValid = (tm < M);               // wave-uniform (M may be 32)
    int row  = tm + r;
    int col0 = tn + r;
    int col1 = tn + 16 + r;
    float s0 = (!NGUARD || col0 < N) ? 1.f : 0.f;
    float s1 = (!NGUARD || col1 < N) ? 1.f : 0.f;

    const float* arow = A + (size_t)(rowsValid ? row : 0) * K;
    v8f acc0 = {}, acc1 = {};

    for (int k0 = 0; k0 < K; k0 += 64) {
        // ---- stage B[k0..k0+63, tn..tn+31] -> shB[col][k] (transposed) ----
        // 64*32 = 2048 elements, 128 threads -> 16 async transfers each.
#pragma unroll
        for (int i = 0; i < 16; ++i) {
            int e  = tid + i * 128;
            int k  = e >> 5;                 // 0..63
            int nl = e & 31;                 // 0..31 (consecutive -> coalesced)
            int gcol = NGUARD ? min(tn + nl, N - 1) : (tn + nl);
            const float* gp = W + (size_t)(k0 + k) * N + gcol;
            lds_f32* lp = (lds_f32*)&shB[nl * LDSTRIDE + k];
            asm volatile("global_load_async_to_lds_b32 %0, %1, off"
                         :: "v"(lp), "v"(gp) : "memory");
        }
        asm volatile("s_wait_asynccnt 0x0" ::: "memory");  // own wave's ASYNCcnt
        __syncthreads();                                   // other waves' stages

        if (rowsValid) {
            if (k0 + 128 < K)
                __builtin_prefetch(arow + k0 + 128, 0, 1); // global_prefetch_b8
            v16h alo = load_afrag(arow + k0, hi);
            v16h ahi = load_afrag(arow + k0 + 32, hi);
            v16h b0l = load_bfrag_lds<NGUARD>(shB + r        * LDSTRIDE, hi, 0,  s0);
            v16h b1l = load_bfrag_lds<NGUARD>(shB + (16 + r) * LDSTRIDE, hi, 0,  s1);
            v16h b0h = load_bfrag_lds<NGUARD>(shB + r        * LDSTRIDE, hi, 32, s0);
            v16h b1h = load_bfrag_lds<NGUARD>(shB + (16 + r) * LDSTRIDE, hi, 32, s1);
            acc0 = __builtin_amdgcn_wmma_f32_16x16x32_f16(false, alo, false, b0l, (short)0, acc0, false, false);
            acc1 = __builtin_amdgcn_wmma_f32_16x16x32_f16(false, alo, false, b1l, (short)0, acc1, false, false);
            acc0 = __builtin_amdgcn_wmma_f32_16x16x32_f16(false, ahi, false, b0h, (short)0, acc0, false, false);
            acc1 = __builtin_amdgcn_wmma_f32_16x16x32_f16(false, ahi, false, b1h, (short)0, acc1, false, false);
        }
        __syncthreads();                                   // before next overwrite
    }

    if (!rowsValid) return;
    // C/D layout: lane -> N = r (+16 for second tile), VGPR v -> M = hi*8 + v
#pragma unroll
    for (int v = 0; v < 8; ++v) {
        int m = tm + hi * 8 + v;
        size_t base = (size_t)m * N;
        if (!NGUARD || col0 < N) {
            float xv = acc0[v];
            if (bias)  xv += bias[col0];
            if (resid) xv += resid[base + col0];
            if (relu)  xv = fmaxf(xv, 0.f);
            C[base + col0] = xv;
        }
        if (!NGUARD || col1 < N) {
            float xv = acc1[v];
            if (bias)  xv += bias[col1];
            if (resid) xv += resid[base + col1];
            if (relu)  xv = fmaxf(xv, 0.f);
            C[base + col1] = xv;
        }
    }
}

// ---------------------------------------------------------------------------
// Elementwise / structural kernels
// ---------------------------------------------------------------------------
__global__ void k_fill0(float* p, long n)
{
    long i = (long)blockIdx.x * blockDim.x + threadIdx.x;
    if (i < n) p[i] = 0.f;
}

// pad columns: dst[rows,dk] = [src[rows,sk] | 0]
__global__ void k_padcols(const float* __restrict__ srcp, float* __restrict__ dstp,
                          int rows, int sk, int dk)
{
    int i = blockIdx.x * blockDim.x + threadIdx.x;
    if (i >= rows * dk) return;
    int rr = i / dk, c = i % dk;
    dstp[i] = (c < sk) ? srcp[(size_t)rr * sk + c] : 0.f;
}

// pad rows: dst[drows,nc] = [src[srows,nc] ; 0]
__global__ void k_padrows(const float* __restrict__ srcp, float* __restrict__ dstp,
                          int srows, int drows, int nc)
{
    int i = blockIdx.x * blockDim.x + threadIdx.x;
    if (i >= drows * nc) return;
    int rr = i / nc;
    dstp[i] = (rr < srows) ? srcp[i] : 0.f;
}

// a1[A,256] = concat(a255[A,255], prelabel[A,1])
__global__ void k_pack_a1(const float* __restrict__ a255, const float* __restrict__ pre,
                          float* __restrict__ a1)
{
    int i = blockIdx.x * blockDim.x + threadIdx.x;
    if (i >= An * Dd) return;
    int r = i / Dd, c = i % Dd;
    a1[i] = (c < Dd - 1) ? a255[(size_t)r * (Dd - 1) + c] : pre[r];
}

// GRU gate fusion: gi = x@Wi+bi, gh = h@Wh+bh already computed by GEMMs
__global__ void k_gru(const float* __restrict__ gi, const float* __restrict__ gh,
                      const float* __restrict__ hprev, float* __restrict__ hout)
{
    int i = blockIdx.x * blockDim.x + threadIdx.x;
    if (i >= An * Dd) return;
    int r = i / Dd, c = i % Dd;
    size_t b3 = (size_t)r * 3 * Dd;
    float ir = gi[b3 + c], iz = gi[b3 + Dd + c], in_ = gi[b3 + 2 * Dd + c];
    float hr = gh[b3 + c], hz = gh[b3 + Dd + c], hn  = gh[b3 + 2 * Dd + c];
    float rg = 1.f / (1.f + expf(-(ir + hr)));
    float z  = 1.f / (1.f + expf(-(iz + hz)));
    float nn = tanhf(in_ + rg * hn);
    hout[i] = (1.f - z) * nn + z * hprev[i];
}

// msg_in[e,:] = h[src[e],:] + bond[e,:]
__global__ void k_gather_add(const float* __restrict__ h, const float* __restrict__ bond,
                             const int* __restrict__ src, float* __restrict__ out)
{
    int i = blockIdx.x * blockDim.x + threadIdx.x;
    if (i >= En * Dd) return;
    int e = i / Dd, c = i % Dd;
    out[i] = h[(size_t)src[e] * Dd + c] + bond[i];
}

// agg[dst[e],:] += msg[e,:]   (segment_sum)
__global__ void k_scatter_add(float* __restrict__ agg, const float* __restrict__ msg,
                              const int* __restrict__ dst)
{
    int i = blockIdx.x * blockDim.x + threadIdx.x;
    if (i >= En * Dd) return;
    int e = i / Dd, c = i % Dd;
    atomicAdd(&agg[(size_t)dst[e] * Dd + c], msg[i]);
}

// MHA core: one block per (batch, head), one thread per query row (N=64, dh=32)
// bias (optional) laid out (B,H,N,N)
__global__ void k_attn(const float* __restrict__ Q, const float* __restrict__ Kv,
                       const float* __restrict__ V, const float* __restrict__ bias,
                       float* __restrict__ O)
{
    int b = blockIdx.x / Hh, h = blockIdx.x % Hh;
    int q = threadIdx.x;
    const float* qp = Q + ((size_t)(b * Nn + q)) * Dd + h * DH;
    float s[Nn];
    float mx = -3.0e38f;
    for (int k = 0; k < Nn; ++k) {
        const float* kp = Kv + ((size_t)(b * Nn + k)) * Dd + h * DH;
        float dot = 0.f;
        for (int d = 0; d < DH; ++d) dot += qp[d] * kp[d];
        dot *= 0.1767766952966369f;     // 1/sqrt(32)
        if (bias) dot += bias[(((size_t)b * Hh + h) * Nn + q) * Nn + k];
        s[k] = dot;
        mx = fmaxf(mx, dot);
    }
    float sum = 0.f;
    for (int k = 0; k < Nn; ++k) { s[k] = expf(s[k] - mx); sum += s[k]; }
    float inv = 1.f / sum;
    for (int d = 0; d < DH; ++d) {
        float o = 0.f;
        for (int k = 0; k < Nn; ++k)
            o += s[k] * V[((size_t)(b * Nn + k)) * Dd + h * DH + d];
        O[((size_t)(b * Nn + q)) * Dd + h * DH + d] = o * inv;
    }
}

// out[r,:] = LN(xin[r,:] + resid[r,:]) * g + b      (one block per row, 256 threads)
__global__ void k_layernorm(const float* __restrict__ xin, const float* __restrict__ resid,
                            const float* __restrict__ g, const float* __restrict__ b,
                            float* __restrict__ out)
{
    __shared__ float sh[Dd];
    int r = blockIdx.x, c = threadIdx.x;
    size_t i = (size_t)r * Dd + c;
    float v = xin[i] + (resid ? resid[i] : 0.f);
    sh[c] = v; __syncthreads();
    for (int s2 = Dd / 2; s2 > 0; s2 >>= 1) { if (c < s2) sh[c] += sh[c + s2]; __syncthreads(); }
    float mean = sh[0] * (1.f / Dd); __syncthreads();
    float d = v - mean;
    sh[c] = d * d; __syncthreads();
    for (int s2 = Dd / 2; s2 > 0; s2 >>= 1) { if (c < s2) sh[c] += sh[c + s2]; __syncthreads(); }
    float var = sh[0] * (1.f / Dd);
    out[i] = g[c] * d * rsqrtf(var + 1e-5f) + b[c];
}

// out[A, 2D] = concat(X[A,D], Y[A,D])
__global__ void k_concat2(const float* __restrict__ X, const float* __restrict__ Y,
                          float* __restrict__ out)
{
    int i = blockIdx.x * blockDim.x + threadIdx.x;
    if (i >= An * 2 * Dd) return;
    int r = i / (2 * Dd), c = i % (2 * Dd);
    out[i] = (c < Dd) ? X[(size_t)r * Dd + c] : Y[(size_t)r * Dd + c - Dd];
}

// bias[(g,h,u,v)] += eb[e,h]   where u=src%N, v=dst%N, g=src/N
__global__ void k_bias_scatter(float* __restrict__ bias, const float* __restrict__ eb,
                               const int* __restrict__ src, const int* __restrict__ dst)
{
    int i = blockIdx.x * blockDim.x + threadIdx.x;
    if (i >= En * Hh) return;
    int e = i / Hh, h = i % Hh;
    int s = src[e], d = dst[e];
    int g = s / Nn, u = s % Nn, v = d % Nn;
    atomicAdd(&bias[(((size_t)g * Hh + h) * Nn + u) * Nn + v], eb[(size_t)e * Hh + h]);
}

// bondcat[BP, 2D] = concat(af[eo0[r],:], af[eo1[r],:])
__global__ void k_gather_pair(const float* __restrict__ af, const int* __restrict__ eo0,
                              const int* __restrict__ eo1, float* __restrict__ out)
{
    int i = blockIdx.x * blockDim.x + threadIdx.x;
    if (i >= BPn * 2 * Dd) return;
    int r = i / (2 * Dd), c = i % (2 * Dd);
    int node = (c < Dd) ? eo0[r] : eo1[r];
    int cc   = (c < Dd) ? c : c - Dd;
    out[i] = af[(size_t)node * Dd + cc];
}

// gmean[b,c] = mean_n af[(b*N+n), c]
__global__ void k_graph_mean(const float* __restrict__ af, float* __restrict__ out)
{
    int i = blockIdx.x * blockDim.x + threadIdx.x;
    if (i >= Bg * Dd) return;
    int b = i / Dd, c = i % Dd;
    float s = 0.f;
    for (int n = 0; n < Nn; ++n) s += af[((size_t)(b * Nn + n)) * Dd + c];
    out[i] = s * (1.f / Nn);
}

// per-graph argmax over bond scores + graph head -> prelabel feedback
__global__ void k_label_update(const float* __restrict__ bond, const float* __restrict__ graph,
                               const int* __restrict__ eo0, const int* __restrict__ eo1,
                               float* __restrict__ pre)
{
    __shared__ float bv[256];
    __shared__ int   bix[256];
    int b = blockIdx.x, t = threadIdx.x;
    const int len = Pp * VOCn;
    float best = -3.0e38f;
    int   bi   = len;
    for (int i = t; i < len; i += 256) {
        float v = bond[(size_t)b * len + i];
        if (v > best) { best = v; bi = i; }     // first-occurrence semantics
    }
    bv[t] = best; bix[t] = bi; __syncthreads();
    for (int s2 = 128; s2 > 0; s2 >>= 1) {
        if (t < s2) {
            if (bv[t + s2] > bv[t] || (bv[t + s2] == bv[t] && bix[t + s2] < bix[t])) {
                bv[t] = bv[t + s2]; bix[t] = bix[t + s2];
            }
        }
        __syncthreads();
    }
    if (t == 0) {
        int idx   = bix[0] / VOCn;
        float inc = (graph[b * 2 + 0] >= graph[b * 2 + 1]) ? 1.f : 0.f;  // argmax==0
        pre[eo0[idx] + b * Nn] += inc;
        pre[eo1[idx] + b * Nn] += inc;
    }
}

// ---------------------------------------------------------------------------
// Host-side helpers (all launches on `stream`, graph-capture safe)
// ---------------------------------------------------------------------------
static inline void gemm(hipStream_t s, const float* A, const float* W, const float* bias,
                        const float* resid, float* C, int M, int K, int N, int relu)
{
    // M % 16 == 0 and K % 64 == 0 guaranteed by padding in kernel_launch
    int tiles = ((M + 63) / 64) * ((N + 31) / 32);
    if ((N & 31) == 0)
        k_gemm_wmma<false><<<dim3(tiles), dim3(128), 0, s>>>(A, W, bias, resid, C, M, K, N, relu);
    else
        k_gemm_wmma<true><<<dim3(tiles), dim3(128), 0, s>>>(A, W, bias, resid, C, M, K, N, relu);
}
static inline void zero(hipStream_t s, float* p, size_t n)
{
    k_fill0<<<dim3((unsigned)((n + 255) / 256)), dim3(256), 0, s>>>(p, (long)n);
}
static inline dim3 grd(size_t n) { return dim3((unsigned)((n + 255) / 256)); }

// ---------------------------------------------------------------------------
// Orchestration.  Parameter pointer indices assume the harness flattens the
// setup_inputs() dict in Python insertion order, with the nested `params`
// dict/list leaves flattened depth-first in insertion order (77 inputs).
// ---------------------------------------------------------------------------
extern "C" void kernel_launch(void* const* d_in, const int* in_sizes, int n_in,
                              void* d_out, int out_size, void* d_ws, size_t ws_size,
                              hipStream_t stream)
{
    (void)in_sizes; (void)n_in; (void)out_size; (void)ws_size;
#define PRM(i) ((const float*)d_in[(i)])
    const float* x  = PRM(0);
    const float* ea = PRM(1);
    const int*   ei = (const int*)d_in[2];
    const int*   eo = (const int*)d_in[3];
    const int* src = ei;
    const int* dst = ei + En;
    const int* eo0 = eo;
    const int* eo1 = eo + BPn;

    const float *Wa1 = PRM(4),  *ba1 = PRM(5),  *Wa2p = PRM(6),  *ba2 = PRM(7);
    const float *Wb1 = PRM(8),  *bb1 = PRM(9),  *Wb2  = PRM(10), *bb2 = PRM(11);
    const float *GWi = PRM(12), *GWh = PRM(13), *Gbi  = PRM(14), *Gbh = PRM(15);
    const float *MWm = PRM(16), *Mbm = PRM(17), *MWh  = PRM(18), *Mbh = PRM(19);
    const float* ATT[2][6];
    for (int l = 0; l < 2; ++l) for (int j = 0; j < 6; ++j) ATT[l][j] = PRM(20 + l * 6 + j);
    const float *A2W1 = PRM(32), *A2b1 = PRM(33), *A2W2 = PRM(34), *A2b2 = PRM(35);
    const float *WbiasP = PRM(36);
    const float* GRF[2][12];   // Wq Wk Wv Wo g b W1f b1f W2f b2f g2 b2
    for (int l = 0; l < 2; ++l) for (int j = 0; j < 12; ++j) GRF[l][j] = PRM(37 + l * 12 + j);
    const float *A3W1 = PRM(61), *A3b1 = PRM(62), *A3W2 = PRM(63), *A3b2 = PRM(64);
    const float *BoW1 = PRM(65), *Bob1 = PRM(66), *BoW2 = PRM(67), *Bob2 = PRM(68);
    const float *BoW3 = PRM(69), *Bob3 = PRM(70);
    const float *GoW1 = PRM(71), *Gob1 = PRM(72), *GoW2 = PRM(73), *Gob2 = PRM(74);
    const float *GoW3 = PRM(75), *Gob3 = PRM(76);
#undef PRM

    float* out = (float*)d_out;

    // ---- workspace bump allocator (floats, 256B aligned blocks) ----
    float* w = (float*)d_ws;
    size_t off = 0;
    auto alloc = [&](size_t n) { float* p = w + off; off += (n + 63) & ~(size_t)63; return p; };

    float* prelabel = alloc(An);
    float* af2      = alloc((size_t)An * Dd);
    float* xpad     = alloc((size_t)An * NAp);
    float* eapad    = alloc((size_t)En * NBp);
    float* Wa1p     = alloc((size_t)NAp * 2 * Dd);
    float* Wb1p     = alloc((size_t)NBp * 2 * Dd);
    float* a1mid    = alloc((size_t)An * 2 * Dd);
    float* a255     = alloc((size_t)An * (Dd - 1));
    float* a1       = alloc((size_t)An * Dd);
    float* bfmid    = alloc((size_t)En * 2 * Dd);
    float* bf       = alloc((size_t)En * Dd);
    float* h0       = alloc((size_t)An * Dd);
    float* h1       = alloc((size_t)An * Dd);
    float* gi       = alloc((size_t)An * 3 * Dd);
    float* gh       = alloc((size_t)An * 3 * Dd);
    float* af       = alloc((size_t)An * Dd);
    float* msg_in   = alloc((size_t)En * Dd);
    float* msg      = alloc((size_t)En * Dd);
    float* agg      = alloc((size_t)An * Dd);
    float* g0b      = alloc((size_t)An * Dd);
    float* g1b      = alloc((size_t)An * Dd);
    float* Qb       = alloc((size_t)An * Dd);
    float* Kb       = alloc((size_t)An * Dd);
    float* Vb       = alloc((size_t)An * Dd);
    float* Ob       = alloc((size_t)An * Dd);
    float* t1b      = alloc((size_t)An * Dd);
    float* t2b      = alloc((size_t)An * Dd);
    float* atb0     = alloc((size_t)An * Dd);
    float* atb1     = alloc((size_t)An * Dd);
    float* cat2     = alloc((size_t)An * 2 * Dd);
    float* big1     = alloc((size_t)An * 4 * Dd);
    float* afm      = alloc((size_t)An * Dd);
    float* ffh      = alloc((size_t)An * 2 * Dd);
    float* ebuf     = alloc((size_t)En * Hh);
    float* biasbuf  = alloc((size_t)Bg * Hh * Nn * Nn);
    float* bondcat  = alloc((size_t)BPn * 2 * Dd);
    float* bondh1   = alloc((size_t)BPn * 2 * Dd);
    float* bondh2   = alloc((size_t)BPn * Dd);
    float* gmean    = alloc((size_t)Bg * Dd);
    float* gm1      = alloc((size_t)Bg * 2 * Dd);
    float* gm2      = alloc((size_t)Bg * Dd);
    float* atb[2] = { atb0, atb1 };

    zero(stream, prelabel, An);
    zero(stream, af2, (size_t)An * Dd);

    // zero-pad embedding weights along K once (rows 98->128, 6->64)
    k_padrows<<<grd((size_t)NAp * 2 * Dd), 256, 0, stream>>>(Wa1, Wa1p, NAa, NAp, 2 * Dd);
    k_padrows<<<grd((size_t)NBp * 2 * Dd), 256, 0, stream>>>(Wb1, Wb1p, NBb, NBp, 2 * Dd);

    for (int t = 0; t < ELn; ++t) {
        const float* xt  = x  + (size_t)t * An * NAa;
        const float* eat = ea + (size_t)t * En * NBb;
        float* bond_out  = out + (size_t)t * Bg * Pp * VOCn;
        float* graph_out = out + (size_t)ELn * Bg * Pp * VOCn + (size_t)t * Bg * 2;

        // pad activations along K (98->128, 6->64)
        k_padcols<<<grd((size_t)An * NAp), 256, 0, stream>>>(xt, xpad, An, NAa, NAp);
        k_padcols<<<grd((size_t)En * NBp), 256, 0, stream>>>(eat, eapad, En, NBb, NBp);

        // node embed: a1 = concat(mlp2(x, w_a), prelabel)
        gemm(stream, xpad, Wa1p, ba1, nullptr, a1mid, An, NAp, 2 * Dd, 1);
        gemm(stream, a1mid, Wa2p, ba2, nullptr, a255, An, 2 * Dd, Dd - 1, 0);
        k_pack_a1<<<grd((size_t)An * Dd), 256, 0, stream>>>(a255, prelabel, a1);

        // edge embed: bf = mlp2(edge_attr, w_b)
        gemm(stream, eapad, Wb1p, bb1, nullptr, bfmid, En, NBp, 2 * Dd, 1);
        gemm(stream, bfmid, Wb2, bb2, nullptr, bf, En, 2 * Dd, Dd, 0);

        // two GRU cells: h = gru(a1, 0); af = gru(af2, h)
        zero(stream, h0, (size_t)An * Dd);
        gemm(stream, a1, GWi, Gbi, nullptr, gi, An, Dd, 3 * Dd, 0);
        gemm(stream, h0, GWh, Gbh, nullptr, gh, An, Dd, 3 * Dd, 0);
        k_gru<<<grd((size_t)An * Dd), 256, 0, stream>>>(gi, gh, h0, h1);
        gemm(stream, af2, GWi, Gbi, nullptr, gi, An, Dd, 3 * Dd, 0);
        gemm(stream, h1, GWh, Gbh, nullptr, gh, An, Dd, 3 * Dd, 0);
        k_gru<<<grd((size_t)An * Dd), 256, 0, stream>>>(gi, gh, h1, af);

        // MPNN (3 rounds)
        const float* hc = af;
        for (int it = 0; it < 3; ++it) {
            k_gather_add<<<grd((size_t)En * Dd), 256, 0, stream>>>(hc, bf, src, msg_in);
            gemm(stream, msg_in, MWm, Mbm, nullptr, msg, En, Dd, Dd, 1);
            zero(stream, agg, (size_t)An * Dd);
            k_scatter_add<<<grd((size_t)En * Dd), 256, 0, stream>>>(agg, msg, dst);
            float* ho = (it & 1) ? g1b : g0b;
            gemm(stream, hc, MWh, Mbh, agg, ho, An, Dd, Dd, 1);
            hc = ho;
        }
        const float* af_gnn = hc;

        // plain attention stack (2 layers, no bias)
        const float* hp = af;
        for (int l = 0; l < 2; ++l) {
            gemm(stream, hp, ATT[l][0], nullptr, nullptr, Qb, An, Dd, Dd, 0);
            gemm(stream, hp, ATT[l][1], nullptr, nullptr, Kb, An, Dd, Dd, 0);
            gemm(stream, hp, ATT[l][2], nullptr, nullptr, Vb, An, Dd, Dd, 0);
            k_attn<<<Bg * Hh, Nn, 0, stream>>>(Qb, Kb, Vb, nullptr, Ob);
            gemm(stream, Ob, ATT[l][3], nullptr, nullptr, t1b, An, Dd, Dd, 0);
            k_layernorm<<<An, Dd, 0, stream>>>(t1b, hp, ATT[l][4], ATT[l][5], atb[l]);
            hp = atb[l];
        }

        // combine: af = mlp2(concat(af_gnn, af_attn), w_a2)
        k_concat2<<<grd((size_t)An * 2 * Dd), 256, 0, stream>>>(af_gnn, hp, cat2);
        gemm(stream, cat2, A2W1, A2b1, nullptr, big1, An, 2 * Dd, 4 * Dd, 1);
        gemm(stream, big1, A2W2, A2b2, nullptr, afm, An, 4 * Dd, Dd, 0);

        // graphormer attention bias
        gemm(stream, bf, WbiasP, nullptr, nullptr, ebuf, En, Dd, Hh, 0);
        zero(stream, biasbuf, (size_t)Bg * Hh * Nn * Nn);
        k_bias_scatter<<<grd((size_t)En * Hh), 256, 0, stream>>>(biasbuf, ebuf, src, dst);

        // graphormer stack (2 layers, biased attention + FFN)
        hp = afm;
        for (int l = 0; l < 2; ++l) {
            gemm(stream, hp, GRF[l][0], nullptr, nullptr, Qb, An, Dd, Dd, 0);
            gemm(stream, hp, GRF[l][1], nullptr, nullptr, Kb, An, Dd, Dd, 0);
            gemm(stream, hp, GRF[l][2], nullptr, nullptr, Vb, An, Dd, Dd, 0);
            k_attn<<<Bg * Hh, Nn, 0, stream>>>(Qb, Kb, Vb, biasbuf, Ob);
            gemm(stream, Ob, GRF[l][3], nullptr, nullptr, t1b, An, Dd, Dd, 0);
            k_layernorm<<<An, Dd, 0, stream>>>(t1b, hp, GRF[l][4], GRF[l][5], t2b);
            gemm(stream, t2b, GRF[l][6], GRF[l][7], nullptr, ffh, An, Dd, 2 * Dd, 1);
            gemm(stream, ffh, GRF[l][8], GRF[l][9], nullptr, t1b, An, 2 * Dd, Dd, 0);
            k_layernorm<<<An, Dd, 0, stream>>>(t1b, t2b, GRF[l][10], GRF[l][11], atb[l]);
            hp = atb[l];
        }
        const float* aff = hp;

        // feedback embedding: af2 = mlp2(aff, w_a3)
        gemm(stream, aff, A3W1, A3b1, nullptr, ffh, An, Dd, 2 * Dd, 1);
        gemm(stream, ffh, A3W2, A3b2, nullptr, af2, An, 2 * Dd, Dd, 0);

        // bond head: mlp3(concat(af[eo0], af[eo1]))
        k_gather_pair<<<grd((size_t)BPn * 2 * Dd), 256, 0, stream>>>(aff, eo0, eo1, bondcat);
        gemm(stream, bondcat, BoW1, Bob1, nullptr, bondh1, BPn, 2 * Dd, 2 * Dd, 1);
        gemm(stream, bondh1, BoW2, Bob2, nullptr, bondh2, BPn, 2 * Dd, Dd, 1);
        gemm(stream, bondh2, BoW3, Bob3, nullptr, bond_out, BPn, Dd, VOCn, 0);

        // graph head: mlp3(mean_n af)
        k_graph_mean<<<grd((size_t)Bg * Dd), 256, 0, stream>>>(aff, gmean);
        gemm(stream, gmean, GoW1, Gob1, nullptr, gm1, Bg, Dd, 2 * Dd, 1);
        gemm(stream, gm1, GoW2, Gob2, nullptr, gm2, Bg, 2 * Dd, Dd, 1);
        gemm(stream, gm2, GoW3, Gob3, nullptr, graph_out, Bg, Dd, 2, 0);

        // argmax feedback into prelabel
        k_label_update<<<Bg, 256, 0, stream>>>(bond_out, graph_out, eo0, eo1, prelabel);
    }
}
